// Corr_89077621719286
// MI455X (gfx1250) — matrix operand order repaired and checked by reading, not compiled, not run on hardware
//
#include <hip/hip_runtime.h>
#include <math.h>

typedef __attribute__((ext_vector_type(16))) __bf16 v16bf;
typedef __attribute__((ext_vector_type(8)))  float  v8f;

__device__ __forceinline__ unsigned short f2bf(float f) {
    unsigned x = __float_as_uint(f);
    unsigned r = (x + 0x7FFFu + ((x >> 16) & 1u)) >> 16;
    return (unsigned short)r;
}
__device__ __forceinline__ float silu(float z) { return z / (1.0f + expf(-z)); }

// ---- softmax over past_clip_ids[:, :3] -> weights [B,3] -------------------
__global__ void k_weights(const float* __restrict__ past, float* __restrict__ wgt) {
    int b = threadIdx.x;
    if (b >= 4) return;
    float v0 = past[b * 4 + 0], v1 = past[b * 4 + 1], v2 = past[b * 4 + 2];
    float m = fmaxf(v0, fmaxf(v1, v2));
    float e0 = expf(v0 - m), e1 = expf(v1 - m), e2 = expf(v2 - m);
    float s = e0 + e1 + e2;
    wgt[b * 3 + 0] = e0 / s; wgt[b * 3 + 1] = e1 / s; wgt[b * 3 + 2] = e2 / s;
}

// ---- channel-normalize f2 into position-major nf [bt, pos, c] -------------
__global__ void k_normalize(const float* __restrict__ f2, float* __restrict__ nf) {
    int idx = blockIdx.x * blockDim.x + threadIdx.x;   // 16*400 threads
    if (idx >= 16 * 400) return;
    int bt = idx / 400, pos = idx % 400;
    const float* src = f2 + (size_t)bt * 512 * 400 + pos;
    float s = 0.f;
    for (int c = 0; c < 512; ++c) { float v = src[(size_t)c * 400]; s += v * v; }
    float inv = 1.0f / fmaxf(sqrtf(s), 1e-12f);
    float* dst = nf + ((size_t)bt * 400 + pos) * 512;
    for (int c = 0; c < 512; ++c) dst[c] = src[(size_t)c * 400] * inv;
}

// ---- raw correlation q[(b,t), d, y, x] = sum_c x1 * shift(x2) -------------
__global__ void k_corr_q(const float* __restrict__ nf, float* __restrict__ q) {
    int p = blockIdx.x;            // pair 0..11
    int d = blockIdx.y;            // offset 0..80
    int b = p / 3, t = p % 3;
    int dy = d / 9 - 4, dx = d % 9 - 4;
    for (int pos = threadIdx.x; pos < 400; pos += blockDim.x) {
        int y = pos / 20, x = pos % 20;
        int y2 = y + dy, x2 = x + dx;
        float dot = 0.f;
        if (y2 >= 0 && y2 < 20 && x2 >= 0 && x2 < 20) {
            const float4* a  = (const float4*)(nf + ((size_t)(b * 4 + t + 1) * 400 + pos) * 512);
            const float4* bb = (const float4*)(nf + ((size_t)(b * 4 + t) * 400 + (y2 * 20 + x2)) * 512);
            for (int c = 0; c < 128; ++c) {
                float4 av = a[c], bv = bb[c];
                dot += av.x * bv.x + av.y * bv.y + av.z * bv.z + av.w * bv.w;
            }
        }
        q[((size_t)p * 81 + d) * 400 + pos] = dot;
    }
}

// ---- 9x9 box filter + softmax over 81 offsets + temporal mix --------------
__global__ void k_box_softmax(const float* __restrict__ q, const float* __restrict__ wgt,
                              float* __restrict__ corr) {
    int b = blockIdx.x;            // 0..3
    int pos = blockIdx.y;          // 0..399
    int y = pos / 20, x = pos % 20;
    int d = threadIdx.x;           // 96 threads, 81 active
    __shared__ float se[81];
    __shared__ float sm, ss;
    float acc = 0.f;
    for (int t = 0; t < 3; ++t) {
        float s = 0.f;
        if (d < 81) {
            const float* qb = q + ((size_t)(b * 3 + t) * 81 + d) * 400;
            for (int i = -4; i <= 4; ++i) {
                int yy = y + i; if (yy < 0 || yy >= 20) continue;
                for (int j = -4; j <= 4; ++j) {
                    int xx = x + j; if (xx < 0 || xx >= 20) continue;
                    s += qb[yy * 20 + xx];
                }
            }
            se[d] = s;
        }
        __syncthreads();
        if (threadIdx.x == 0) { float m = se[0]; for (int k = 1; k < 81; ++k) m = fmaxf(m, se[k]); sm = m; }
        __syncthreads();
        float e = 0.f;
        if (d < 81) { e = expf(s - sm); se[d] = e; }
        __syncthreads();
        if (threadIdx.x == 0) { float tt = 0.f; for (int k = 0; k < 81; ++k) tt += se[k]; ss = tt; }
        __syncthreads();
        if (d < 81) acc += wgt[b * 3 + t] * e / ss;
        __syncthreads();
    }
    if (d < 81) corr[((size_t)b * 81 + d) * 400 + pos] = acc;
}

// ---- f32 weight matrix -> bf16 [rows, dstCols], zero-padded ---------------
__global__ void k_stage_w(const float* __restrict__ src, unsigned short* __restrict__ dst,
                          int rows, int srcCols, int srcOff, int dstCols, int copyCols) {
    int idx = blockIdx.x * blockDim.x + threadIdx.x;
    if (idx >= rows * dstCols) return;
    int r = idx / dstCols, j = idx % dstCols;
    float v = (j < copyCols) ? src[(size_t)r * srcCols + srcOff + j] : 0.f;
    dst[idx] = f2bf(v);
}

// ---- f[:, -1] -> position-major bf16 Xa [(b*HW+pos), c] -------------------
__global__ void k_stage_xa(const float* __restrict__ f, unsigned short* __restrict__ Xa,
                           int C, int HW) {
    int idx = blockIdx.x * blockDim.x + threadIdx.x;
    if (idx >= 4 * HW * C) return;
    int c = idx % C; int rem = idx / C; int pos = rem % HW; int b = rem / HW;
    float v = f[(((size_t)b * 4 + 3) * C + c) * HW + pos];
    Xa[idx] = f2bf(v);
}

// ---- bilinear resize (align_corners) corr -> bf16 [(b*HW+pos), 96] --------
__global__ void k_resize(const float* __restrict__ corr, unsigned short* __restrict__ cr,
                         int H, int W) {
    int idx = blockIdx.x * blockDim.x + threadIdx.x;
    if (idx >= 4 * H * W * 96) return;
    int d = idx % 96; int rem = idx / 96; int pos = rem % (H * W); int b = rem / (H * W);
    float v = 0.f;
    if (d < 81) {
        int y = pos / W, x = pos % W;
        float fy = (H == 20) ? (float)y : (float)y * (19.0f / (float)(H - 1));
        float fx = (W == 20) ? (float)x : (float)x * (19.0f / (float)(W - 1));
        int y0 = (int)floorf(fy); int x0 = (int)floorf(fx);
        int y1 = min(y0 + 1, 19), x1 = min(x0 + 1, 19);
        float ty = fy - (float)y0, tx = fx - (float)x0;
        const float* cb = corr + ((size_t)b * 81 + d) * 400;
        float v00 = cb[y0 * 20 + x0], v01 = cb[y0 * 20 + x1];
        float v10 = cb[y1 * 20 + x0], v11 = cb[y1 * 20 + x1];
        v = (1.f - ty) * ((1.f - tx) * v00 + tx * v01) + ty * ((1.f - tx) * v10 + tx * v11);
    }
    cr[idx] = f2bf(v);
}

// ---- combine conv1 parts + affine + SiLU -> bf16 y1 [(b,tf,pos), o] -------
__global__ void k_y1(const float* __restrict__ A, const float* __restrict__ Cc,
                     const float* __restrict__ fut, const float* __restrict__ g1,
                     const float* __restrict__ b1, unsigned short* __restrict__ y1,
                     int C, int HW) {
    int idx = blockIdx.x * blockDim.x + threadIdx.x;
    if (idx >= 16 * HW * C) return;
    int o = idx % C; int rem = idx / C; int pos = rem % HW;
    int rem2 = rem / HW; int tf = rem2 % 4; int b = rem2 / 4;
    size_t r = ((size_t)b * HW + pos) * C + o;
    float fid = fut[b * 4 + tf];
    float z = g1[o] * (A[r] + fid * Cc[r]) + b1[o];
    y1[idx] = f2bf(silu(z));
}

// ---- WMMA bf16 GEMM: D[m,n] = sum_k X[m,k] * Wm[n,k] ----------------------
// 2x2 register blocking: each wave computes a 32x32 output tile (4 wmma acc).
// Block = 4 waves tiled 2x2 over (M,N) -> block tile 64x64.
// Requires M % 64 == 0, N % 64 == 0, K % 32 == 0 (true for all call sites).
// Fragments loaded as explicit uint4 (guaranteed global_load_b128):
//   A 16x32 bf16: lane row = lane&15; two aligned 4-uint runs at kh+hi4,
//   kh+8+hi4 (K-pairs 2v(+8 v>=4)(+8 hi)).  B 32x16: lane col = lane&15;
//   runs at kh+hi4, kh+8+hi4 (same K order as A).
// EP=0: store f32 row-major [m,N]. EP=1: affine+SiLU, channel-major output.
template <int EP>
__global__ void k_gemm(const unsigned short* __restrict__ X,
                       const unsigned short* __restrict__ Wm,
                       float* __restrict__ D, int N, int K, int HW,
                       const float* __restrict__ g, const float* __restrict__ bias) {
    int lane = threadIdx.x & 31;
    int wave = threadIdx.x >> 5;          // 0..3
    int wm = wave >> 1, wn = wave & 1;    // 2x2 wave grid
    int mBase = blockIdx.x * 64 + wm * 32;
    int nBase = blockIdx.y * 64 + wn * 32;
    int ln = lane & 15;
    int hi = (lane >> 4) & 1;
    int hi4 = hi * 4, hi8 = hi * 8;

    const unsigned* xrow0 = (const unsigned*)(X + (size_t)(mBase + ln) * K);
    const unsigned* xrow1 = (const unsigned*)(X + (size_t)(mBase + 16 + ln) * K);
    const unsigned* wrow0 = (const unsigned*)(Wm + (size_t)(nBase + ln) * K);
    const unsigned* wrow1 = (const unsigned*)(Wm + (size_t)(nBase + 16 + ln) * K);

    // One-shot speculative prefetch (global_prefetch_b8); keeps the K-loop
    // branch-free so addressing stays strength-reduced with const offsets.
    __builtin_prefetch(xrow0, 0, 1);
    __builtin_prefetch(xrow1, 0, 1);
    __builtin_prefetch(wrow0, 0, 1);
    __builtin_prefetch(wrow1, 0, 1);

    v8f acc00 = {0.f, 0.f, 0.f, 0.f, 0.f, 0.f, 0.f, 0.f};
    v8f acc01 = acc00, acc10 = acc00, acc11 = acc00;
    union FragU { v16bf v; uint4 q[2]; };
    FragU a0, a1, b0, b1;

    for (int k0 = 0; k0 < K; k0 += 32) {
        int kh = k0 >> 1;  // index in 32-bit (bf16-pair) units
        a0.q[0] = *(const uint4*)(xrow0 + kh + hi4);
        a0.q[1] = *(const uint4*)(xrow0 + kh + 8 + hi4);
        a1.q[0] = *(const uint4*)(xrow1 + kh + hi4);
        a1.q[1] = *(const uint4*)(xrow1 + kh + 8 + hi4);
        b0.q[0] = *(const uint4*)(wrow0 + kh + hi4);
        b0.q[1] = *(const uint4*)(wrow0 + kh + 8 + hi4);
        b1.q[0] = *(const uint4*)(wrow1 + kh + hi4);
        b1.q[1] = *(const uint4*)(wrow1 + kh + 8 + hi4);
        acc00 = __builtin_amdgcn_wmma_f32_16x16x32_bf16(false, a0.v, false, b0.v, (short)0, acc00, false, false);
        acc01 = __builtin_amdgcn_wmma_f32_16x16x32_bf16(false, a0.v, false, b1.v, (short)0, acc01, false, false);
        acc10 = __builtin_amdgcn_wmma_f32_16x16x32_bf16(false, a1.v, false, b0.v, (short)0, acc10, false, false);
        acc11 = __builtin_amdgcn_wmma_f32_16x16x32_bf16(false, a1.v, false, b1.v, (short)0, acc11, false, false);
    }

#pragma unroll
    for (int mi = 0; mi < 2; ++mi) {
#pragma unroll
        for (int ni = 0; ni < 2; ++ni) {
            v8f acc = (mi == 0) ? (ni == 0 ? acc00 : acc01) : (ni == 0 ? acc10 : acc11);
            int mT = mBase + mi * 16;
            int nT = nBase + ni * 16;
#pragma unroll
            for (int v = 0; v < 8; ++v) {
                int m = mT + v + hi * 8;
                int n = nT + ln;
                if (EP == 0) {
                    D[(size_t)m * N + n] = acc[v];
                } else {
                    float z = g[n] * acc[v] + bias[n];
                    float s = silu(z);
                    int bt = m / HW; int pos = m - bt * HW;
                    D[((size_t)bt * N + n) * HW + pos] = s;
                }
            }
        }
    }
}

static inline dim3 g1d(size_t n, int b) { return dim3((unsigned)((n + (size_t)b - 1) / (size_t)b)); }

extern "C" void kernel_launch(void* const* d_in, const int* in_sizes, int n_in,
                              void* d_out, int out_size, void* d_ws, size_t ws_size,
                              hipStream_t stream) {
    const float* f[3] = {(const float*)d_in[0], (const float*)d_in[1], (const float*)d_in[2]};
    const float* past = (const float*)d_in[3];
    const float* fut  = (const float*)d_in[4];
    float* out = (float*)d_out;

    // Workspace carve (sequential reuse across scales; peak ~80 MB)
    char* p = (char*)d_ws;
    auto carve = [&](size_t bytes) -> void* {
        void* r = (void*)p; p += (bytes + 255) & ~(size_t)255; return r;
    };
    float*          wgt  = (float*)carve(12 * 4);
    float*          nf   = (float*)carve((size_t)16 * 400 * 512 * 4);
    float*          q    = (float*)carve((size_t)12 * 81 * 400 * 4);
    float*          corr = (float*)carve((size_t)4 * 81 * 400 * 4);
    unsigned short* Xa   = (unsigned short*)carve((size_t)4 * 6400 * 128 * 2);
    unsigned short* crF  = (unsigned short*)carve((size_t)4 * 6400 * 96 * 2);
    unsigned short* w1a  = (unsigned short*)carve((size_t)512 * 512 * 2);
    unsigned short* w1c  = (unsigned short*)carve((size_t)512 * 96 * 2);
    unsigned short* w2b  = (unsigned short*)carve((size_t)512 * 512 * 2);
    float*          Abuf = (float*)carve((size_t)4 * 6400 * 128 * 4);
    float*          Cbuf = (float*)carve((size_t)4 * 6400 * 128 * 4);
    unsigned short* y1   = (unsigned short*)carve((size_t)16 * 6400 * 128 * 2);

    // Correlation pipeline (shared across scales)
    k_weights<<<1, 32, 0, stream>>>(past, wgt);
    k_normalize<<<25, 256, 0, stream>>>(f[2], nf);
    k_corr_q<<<dim3(12, 81), 256, 0, stream>>>(nf, q);
    k_box_softmax<<<dim3(4, 400), 96, 0, stream>>>(q, wgt, corr);

    const int chans[3] = {128, 256, 512};
    const int hs[3]    = {80, 40, 20};
    size_t outOff = 0;
    for (int i = 0; i < 3; ++i) {
        int C = chans[i], H = hs[i], HW = H * H;
        const float* w1 = (const float*)d_in[5 + 6 * i];
        const float* g1 = (const float*)d_in[6 + 6 * i];
        const float* b1 = (const float*)d_in[7 + 6 * i];
        const float* w2 = (const float*)d_in[8 + 6 * i];
        const float* g2 = (const float*)d_in[9 + 6 * i];
        const float* b2 = (const float*)d_in[10 + 6 * i];
        int KW1 = C + 81;

        k_stage_w<<<g1d((size_t)C * C, 256), 256, 0, stream>>>(w1, w1a, C, KW1, 0, C, C);
        k_stage_w<<<g1d((size_t)C * 96, 256), 256, 0, stream>>>(w1, w1c, C, KW1, C, 96, 81);
        k_stage_w<<<g1d((size_t)C * C, 256), 256, 0, stream>>>(w2, w2b, C, C, 0, C, C);
        k_stage_xa<<<g1d((size_t)4 * HW * C, 256), 256, 0, stream>>>(f[i], Xa, C, HW);
        k_resize<<<g1d((size_t)4 * HW * 96, 256), 256, 0, stream>>>(corr, crF, H, H);

        int M1 = 4 * HW;   // conv1 GEMMs: once per b (tf-factored)
        k_gemm<0><<<dim3(M1 / 64, C / 64), 128, 0, stream>>>(Xa, w1a, Abuf, C, C, 1, nullptr, nullptr);
        k_gemm<0><<<dim3(M1 / 64, C / 64), 128, 0, stream>>>(crF, w1c, Cbuf, C, 96, 1, nullptr, nullptr);
        k_y1<<<g1d((size_t)16 * HW * C, 256), 256, 0, stream>>>(Abuf, Cbuf, fut, g1, b1, y1, C, HW);

        int M2 = 16 * HW;  // conv2 GEMM, fused affine+SiLU, channel-major out
        k_gemm<1><<<dim3(M2 / 64, C / 64), 128, 0, stream>>>(y1, w2b, out + outOff, C, C, HW, g2, b2);
        outOff += (size_t)16 * C * HW;
    }
}
